// CausalSelfAttentionSH_43911745634977
// MI455X (gfx1250) — matrix operand order repaired and checked by reading, not compiled, not run on hardware
//
#include <hip/hip_runtime.h>
#include <hip/hip_bf16.h>

// CausalSelfAttention  B=4 S=4096 E=1024 D=64, fp32 I/O.
// bf16 WMMA pipeline: wprep (w -> w^T bf16) -> qkv_proj (x @ W with async-LDS
// double-buffered weight staging) -> flash_attn (online-softmax, WMMA QK^T/PV).

#define BB 4
#define SS 4096
#define EE 1024
#define DD 64

typedef __attribute__((ext_vector_type(16))) __bf16 v16bf;
typedef __attribute__((ext_vector_type(2)))  __bf16 v2bf;
typedef __attribute__((ext_vector_type(8)))  float  v8f;
typedef int vi4 __attribute__((vector_size(16)));      // b128 payload type

#if __has_builtin(__builtin_amdgcn_global_load_async_to_lds_b128)
#define QKV_ASYNC 1
#else
#define QKV_ASYNC 0
#endif

// builtin wants: (global int4* src, local int4* dst, imm offset, imm cpol)
#define GV4(p) ((__attribute__((address_space(1))) vi4*)(p))
#define LV4(p) ((__attribute__((address_space(3))) vi4*)(p))

static __device__ __forceinline__ v2bf ld_bf16x2(const __bf16* p) {
  unsigned u = *reinterpret_cast<const unsigned*>(p);   // k, k+1 contiguous, 4B aligned
  return __builtin_bit_cast(v2bf, u);
}

// ---------------------------------------------------------------------------
// Stage 0: weight transpose + f32->bf16.  w:[E,D] f32 -> wt:[D,E] bf16 so that
// WMMA B-tiles (lane = column n, VGPRs = K pairs) load as contiguous pairs.
// ---------------------------------------------------------------------------
__global__ void wprep(const float* __restrict__ w, __bf16* __restrict__ wt) {
  int idx = blockIdx.x * blockDim.x + threadIdx.x;      // 0 .. E*D-1
  int e = idx / DD, d = idx - e * DD;
  wt[(size_t)d * EE + e] = (__bf16)w[idx];
}

// ---------------------------------------------------------------------------
// Stage 1: Q/K/V projection.  8 waves/WG, one 16-row tile each.  Per 32-wide
// k-step the WG stages 3x(64x32) bf16 weight chunks (12 KB) into LDS with
// async global->LDS copies, double buffered; waves consume via ds_load_b128.
// Q,K stored row-major bf16 [B*S, D]; V stored transposed bf16 [B, D, S].
// ---------------------------------------------------------------------------
#define CHUNK_BYTES 12288                                // 3 mats * 64 * 64B

static __device__ __forceinline__ void stage_chunk(
    const __bf16* __restrict__ wtq, const __bf16* __restrict__ wtk,
    const __bf16* __restrict__ wtv, char* lbuf, int e, int tid) {
  const int n = tid >> 2, seg = tid & 3;                 // row 0..63, 16B segment
  const size_t goff = (size_t)n * EE + e + seg * 8;      // bf16 elements
  char* l0 = lbuf + n * 64 + seg * 16;
#if QKV_ASYNC
  __builtin_amdgcn_global_load_async_to_lds_b128(GV4(wtq + goff), LV4(l0),        0, 0);
  __builtin_amdgcn_global_load_async_to_lds_b128(GV4(wtk + goff), LV4(l0 + 4096), 0, 0);
  __builtin_amdgcn_global_load_async_to_lds_b128(GV4(wtv + goff), LV4(l0 + 8192), 0, 0);
#else
  *(uint4*)(l0)        = *(const uint4*)(wtq + goff);
  *(uint4*)(l0 + 4096) = *(const uint4*)(wtk + goff);
  *(uint4*)(l0 + 8192) = *(const uint4*)(wtv + goff);
#endif
}

static __device__ __forceinline__ void wait_async_le3() {
#if QKV_ASYNC
#if __has_builtin(__builtin_amdgcn_s_wait_asynccnt)
  __builtin_amdgcn_s_wait_asynccnt(3);
#else
  asm volatile("s_wait_asynccnt 0x3" ::: "memory");
#endif
#endif
}
static __device__ __forceinline__ void wait_async_le0() {
#if QKV_ASYNC
#if __has_builtin(__builtin_amdgcn_s_wait_asynccnt)
  __builtin_amdgcn_s_wait_asynccnt(0);
#else
  asm volatile("s_wait_asynccnt 0x0" ::: "memory");
#endif
#endif
}

__global__ __launch_bounds__(256) void qkv_proj(
    const float* __restrict__ x,
    const __bf16* __restrict__ wtq, const __bf16* __restrict__ wtk,
    const __bf16* __restrict__ wtv,
    __bf16* __restrict__ qb, __bf16* __restrict__ kb, __bf16* __restrict__ vtb) {
  __shared__ __align__(32) char wsm[2 * CHUNK_BYTES];    // 24 KB double buffer
  const int tid  = threadIdx.x;
  const int lane = tid & 31;
  const int wave = tid >> 5;
  const int tile = wave * 128 + blockIdx.x;              // 1024 tiles, strided
  const int row0 = tile * 16;
  const int m    = lane & 15;
  const int kh   = lane >> 4;

  // A-matrix (16-bit, 16x32) K offsets per VGPR pair (contiguous: k = 16*kh + 2v)
  int koffA[8];
#pragma unroll
  for (int v = 0; v < 8; ++v) koffA[v] = (kh << 4) + (v << 1);

  v8f accQ[4] = {}, accK[4] = {}, accV[4] = {};
  const float2* xr = reinterpret_cast<const float2*>(x + (size_t)(row0 + m) * EE);

  stage_chunk(wtq, wtk, wtv, wsm, 0, tid);
  int ib = 0;
  for (int e = 0; e < EE; e += 32) {
    const char* cur = wsm + ib * CHUNK_BYTES;
    if (e + 32 < EE) {
      stage_chunk(wtq, wtk, wtv, wsm + (ib ^ 1) * CHUNK_BYTES, e + 32, tid);
      wait_async_le3();                                  // in-order: current chunk done
    } else {
      wait_async_le0();
    }
    __syncthreads();                                     // chunk visible WG-wide

    v16bf a;
#pragma unroll
    for (int v = 0; v < 8; ++v) {
      float2 f = xr[(e + koffA[v]) >> 1];
      a[2 * v] = (__bf16)f.x; a[2 * v + 1] = (__bf16)f.y;
    }
#pragma unroll
    for (int t = 0; t < 4; ++t) {
      const char* pb = cur + (t * 16 + m) * 64 + (kh << 5);  // 32B aligned
      v16bf bq = *reinterpret_cast<const v16bf*>(pb);
      v16bf bk = *reinterpret_cast<const v16bf*>(pb + 4096);
      v16bf bv = *reinterpret_cast<const v16bf*>(pb + 8192);
      accQ[t] = __builtin_amdgcn_wmma_f32_16x16x32_bf16(false, a, false, bq, (short)0, accQ[t], false, false);
      accK[t] = __builtin_amdgcn_wmma_f32_16x16x32_bf16(false, a, false, bk, (short)0, accK[t], false, false);
      accV[t] = __builtin_amdgcn_wmma_f32_16x16x32_bf16(false, a, false, bv, (short)0, accV[t], false, false);
    }
    __syncthreads();                                     // all reads done before overwrite
    ib ^= 1;
  }

  // C/D layout: VGPR j, lane -> row = j + 8*(lane>=16), col = lane&15
#pragma unroll
  for (int t = 0; t < 4; ++t)
#pragma unroll
    for (int j = 0; j < 8; ++j) {
      const int r = row0 + j + (kh << 3);
      const int c = t * 16 + m;
      qb[(size_t)r * DD + c] = (__bf16)accQ[t][j];
      kb[(size_t)r * DD + c] = (__bf16)accK[t][j];
      const int b = r >> 12, s = r & (SS - 1);
      vtb[((size_t)b * DD + c) * SS + s] = (__bf16)accV[t][j];
    }
}

// ---------------------------------------------------------------------------
// Stage 2: flash attention, one wave per 16-query tile, 32-wide KV blocks.
// logits = (Q K^T) masked(-inf) then *0.125 then softmax (matches reference).
// P transposed C-layout -> A-layout through per-wave LDS (s_wait_dscnt).
// ---------------------------------------------------------------------------
__global__ __launch_bounds__(256) void flash_attn(
    const __bf16* __restrict__ qb, const __bf16* __restrict__ kb,
    const __bf16* __restrict__ vtb, float* __restrict__ out) {
  __shared__ unsigned short pbuf[8][16 * 32];           // 1KB per wave
  const int lane = threadIdx.x & 31;
  const int wave = threadIdx.x >> 5;
  const int qt   = wave * 128 + blockIdx.x;             // strided: balance causal work
  const int r0   = qt * 16;                             // flattened b*S + s
  const int b    = r0 >> 12;
  const int q0   = r0 & (SS - 1);
  const int m    = lane & 15;
  const int kh   = lane >> 4;
  unsigned short* lds = pbuf[wave];

  int koffA[8];
#pragma unroll
  for (int v = 0; v < 8; ++v) koffA[v] = (kh << 4) + (v << 1);

  // Q tile: 16x64 as two 16x32 A-matrices
  v16bf qa[2];
  {
    const __bf16* qr = qb + (size_t)(r0 + m) * DD;
#pragma unroll
    for (int c = 0; c < 2; ++c)
#pragma unroll
      for (int v = 0; v < 8; ++v) {
        v2bf p = ld_bf16x2(&qr[c * 32 + koffA[v]]);
        qa[c][2 * v] = p[0]; qa[c][2 * v + 1] = p[1];
      }
  }

  v8f o[4] = {};
  float mrow[8], lrow[8];
#pragma unroll
  for (int j = 0; j < 8; ++j) { mrow[j] = -INFINITY; lrow[j] = 0.f; }

  const int nkv = q0 + 16;                              // exclusive causal col bound
  for (int kv0 = 0; kv0 < nkv; kv0 += 32) {
    if (kv0 + 32 < nkv) {                               // prefetch next K/V block
      __builtin_prefetch(kb + ((size_t)b * SS + kv0 + 32 + m) * DD, 0, 0);
      __builtin_prefetch(vtb + ((size_t)b * DD + m) * SS + kv0 + 32, 0, 0);
    }
    // ---- S = Q K^T : 2 col-tiles x 2 k-chunks -> 4 WMMA
    v8f sc[2] = {};
#pragma unroll
    for (int t = 0; t < 2; ++t) {
      const __bf16* kr = kb + ((size_t)b * SS + kv0 + t * 16 + m) * DD;
#pragma unroll
      for (int c = 0; c < 2; ++c) {
        v16bf kt;
#pragma unroll
        for (int v = 0; v < 8; ++v) {
          v2bf p = ld_bf16x2(&kr[c * 32 + (kh << 4) + (v << 1)]);
          kt[2 * v] = p[0]; kt[2 * v + 1] = p[1];
        }
        sc[t] = __builtin_amdgcn_wmma_f32_16x16x32_bf16(false, qa[c], false, kt, (short)0, sc[t], false, false);
      }
    }
    // ---- online softmax (row = j + 8*kh across lanes' col = m)
    float p0[8], p1[8];
#pragma unroll
    for (int j = 0; j < 8; ++j) {
      const int grow = q0 + j + (kh << 3);
      float s0 = (kv0 + m      <= grow) ? sc[0][j] * 0.125f : -INFINITY;
      float s1 = (kv0 + 16 + m <= grow) ? sc[1][j] * 0.125f : -INFINITY;
      float mx = fmaxf(s0, s1);
      mx = fmaxf(mx, __shfl_xor(mx, 1));
      mx = fmaxf(mx, __shfl_xor(mx, 2));
      mx = fmaxf(mx, __shfl_xor(mx, 4));
      mx = fmaxf(mx, __shfl_xor(mx, 8));
      const float mnew  = fmaxf(mrow[j], mx);
      const float alpha = __expf(mrow[j] - mnew);
      mrow[j] = mnew;
      float e0 = __expf(s0 - mnew), e1 = __expf(s1 - mnew);
      p0[j] = e0; p1[j] = e1;
      float rs = e0 + e1;
      rs += __shfl_xor(rs, 1);
      rs += __shfl_xor(rs, 2);
      rs += __shfl_xor(rs, 4);
      rs += __shfl_xor(rs, 8);
      lrow[j] = lrow[j] * alpha + rs;
#pragma unroll
      for (int t = 0; t < 4; ++t) o[t][j] = o[t][j] * alpha;
    }
    // ---- transpose P (C-layout -> A-layout) through LDS
#pragma unroll
    for (int j = 0; j < 8; ++j) {
      const int rl = j + (kh << 3);
      lds[rl * 32 + m]      = __builtin_bit_cast(unsigned short, (__bf16)p0[j]);
      lds[rl * 32 + 16 + m] = __builtin_bit_cast(unsigned short, (__bf16)p1[j]);
    }
    asm volatile("s_wait_dscnt 0x0" ::: "memory");
    v16bf pa;
#pragma unroll
    for (int v = 0; v < 8; ++v) {
      unsigned u = *reinterpret_cast<const unsigned*>(&lds[m * 32 + koffA[v]]);
      v2bf p = __builtin_bit_cast(v2bf, u);
      pa[2 * v] = p[0]; pa[2 * v + 1] = p[1];
    }
    // ---- O += P V : 4 col-tiles of D, V^T rows contiguous -> 4 WMMA
#pragma unroll
    for (int t = 0; t < 4; ++t) {
      const __bf16* vr = vtb + ((size_t)b * DD + t * 16 + m) * SS + kv0;
      v16bf vt;
#pragma unroll
      for (int v = 0; v < 8; ++v) {
        v2bf p = ld_bf16x2(&vr[(kh << 4) + (v << 1)]);
        vt[2 * v] = p[0]; vt[2 * v + 1] = p[1];
      }
      o[t] = __builtin_amdgcn_wmma_f32_16x16x32_bf16(false, pa, false, vt, (short)0, o[t], false, false);
    }
  }

  // ---- epilogue: O / l, fp32 out [B,S,D]
#pragma unroll
  for (int t = 0; t < 4; ++t)
#pragma unroll
    for (int j = 0; j < 8; ++j) {
      const int r = r0 + j + (kh << 3);
      out[(size_t)r * DD + t * 16 + m] = o[t][j] / lrow[j];
    }
}

// ---------------------------------------------------------------------------
extern "C" void kernel_launch(void* const* d_in, const int* in_sizes, int n_in,
                              void* d_out, int out_size, void* d_ws, size_t ws_size,
                              hipStream_t stream) {
  const float* x  = (const float*)d_in[0];
  const float* wq = (const float*)d_in[1];
  const float* wk = (const float*)d_in[2];
  const float* wv = (const float*)d_in[3];
  float* out = (float*)d_out;

  char* ws = (char*)d_ws;
  const size_t QKV = (size_t)BB * SS * DD * sizeof(__bf16);   // 2 MB each
  const size_t WT  = (size_t)EE * DD * sizeof(__bf16);        // 128 KB each
  __bf16* qb  = (__bf16*)(ws);
  __bf16* kb  = (__bf16*)(ws + QKV);
  __bf16* vtb = (__bf16*)(ws + 2 * QKV);
  __bf16* wtq = (__bf16*)(ws + 3 * QKV);
  __bf16* wtk = (__bf16*)(ws + 3 * QKV + WT);
  __bf16* wtv = (__bf16*)(ws + 3 * QKV + 2 * WT);

  wprep<<<(EE * DD) / 256, 256, 0, stream>>>(wq, wtq);
  wprep<<<(EE * DD) / 256, 256, 0, stream>>>(wk, wtk);
  wprep<<<(EE * DD) / 256, 256, 0, stream>>>(wv, wtv);
  qkv_proj<<<128, 256, 0, stream>>>(x, wtq, wtk, wtv, qb, kb, vtb);
  flash_attn<<<128, 256, 0, stream>>>(qb, kb, vtb, out);
}
